// InvariantPointAttention_75608604278963
// MI455X (gfx1250) — compile-verified, matrix operand order checked by reading, not run on hardware
//
#include <hip/hip_runtime.h>
#include <hip/hip_bf16.h>
#include <math.h>

// ---------------------------------------------------------------------------
// Invariant Point Attention for MI455X (gfx1250), wave32 + WMMA f16->f32.
// B=1, L=1024, DIM=384, H=12, DH=16, PQ=4, PV=8.
// ---------------------------------------------------------------------------

typedef __attribute__((ext_vector_type(16))) _Float16 v16h;
typedef __attribute__((ext_vector_type(8)))  _Float16 v8h;
typedef __attribute__((ext_vector_type(8)))  float    v8f;

#define L_SEQ 1024
#define DIM   384
#define NHEAD 12
#define DH    16
#define PQ    4
#define PV    8

// ---------------------------------------------------------------------------
// Load a 16x32 f16 WMMA operand fragment (A layout; B uses the mirrored
// layout, so pass the transposed matrix with rows = N).
// Source is row-major: base[row + m][k], row stride = strideHalfs halfs.
// Per ISA 16-bit A layout:
//   lane<16 : M=lane,    K in {k0+0..7,  k0+16..23}
//   lane>=16: M=lane-16, K in {k0+8..15, k0+24..31}
// Each chunk is 8 contiguous halfs = one 16-byte load.
// ---------------------------------------------------------------------------
__device__ __forceinline__ v16h load_frag(const _Float16* base, int row,
                                          int strideHalfs, int k0) {
  const int lane = threadIdx.x & 31;
  const int m    = lane & 15;
  const int hi   = lane >> 4;
  const _Float16* p = base + (size_t)(row + m) * strideHalfs + k0 + hi * 8;
  v8h lo = *(const v8h*)(p);
  v8h hiv = *(const v8h*)(p + 16);
  v16h r;
#pragma unroll
  for (int i = 0; i < 8; ++i) { r[i] = lo[i]; r[8 + i] = hiv[i]; }
  return r;
}

// ---------------------------------------------------------------------------
// Elementwise packs
// ---------------------------------------------------------------------------
__global__ void cvt_f32_to_f16(const float* __restrict__ src,
                               _Float16* __restrict__ dst, int n) {
  int i = blockIdx.x * 256 + threadIdx.x;
  if (i < n) dst[i] = (_Float16)src[i];
}

// w is [K][N] row-major f32 -> wt [N][K] row-major f16
__global__ void transpose_to_f16(const float* __restrict__ w,
                                 _Float16* __restrict__ wt, int K, int N) {
  int i = blockIdx.x * 256 + threadIdx.x;
  if (i < K * N) {
    int k = i / N, n = i % N;
    wt[(size_t)n * K + k] = (_Float16)w[i];
  }
}

// ---------------------------------------------------------------------------
// Generic WMMA GEMM: C[M x N] = A[M x K] * Bt[N x K]^T + bias
// A, Bt row-major f16, K % 32 == 0, one wave per 16x16 tile.
// ---------------------------------------------------------------------------
__global__ void __launch_bounds__(32)
gemm_wmma_f16f32(const _Float16* __restrict__ A, const _Float16* __restrict__ Bt,
                 const float* __restrict__ bias, float* __restrict__ C,
                 int N, int K) {
  const int m0 = blockIdx.x * 16;
  const int n0 = blockIdx.y * 16;
  const int lane = threadIdx.x & 31;
  v8f acc = {};
  for (int k = 0; k < K; k += 32) {
    v16h a = load_frag(A, m0, K, k);
    v16h b = load_frag(Bt, n0, K, k);
    acc = __builtin_amdgcn_wmma_f32_16x16x32_f16(false, a, false, b,
                                                 (short)0, acc, false, false);
  }
  const int hi = lane >> 4;
  const int nn = lane & 15;
  const int col = n0 + nn;
  const float bv = bias[col];
#pragma unroll
  for (int r = 0; r < 8; ++r) {
    int row = m0 + r + 8 * hi;
    C[(size_t)row * N + col] = acc[r] + bv;
  }
}

// ---------------------------------------------------------------------------
// Epilogue of the projections: build attention operand packs.
//  Qpack[h][t][64] = [ q_s*0.25 (16) | 0 (16) | q_p+coords (12) | 0 (20) ]  f16
//  Kpack[h][t][64] = [ k_s       (16)| 0 (16) | k_p+coords (12) | 0 (20) ]  f16
//  Vt[h][48][1024] : cols 0..15 = v_s, 16..39 = v_p, 40..47 = 0           f16
//  q2/k2[h][t] = sum over 12 point dims of (q_p/k_p)^2                     f32
//  upd_coords[t][3] = mean over (h,p) of v_p                               f32
// ---------------------------------------------------------------------------
__global__ void __launch_bounds__(64)
ipa_pack_epilogue(const float* __restrict__ proj_qs, const float* __restrict__ proj_ks,
                  const float* __restrict__ proj_vs, const float* __restrict__ proj_qp,
                  const float* __restrict__ proj_kp, const float* __restrict__ proj_vp,
                  const float* __restrict__ coords,
                  _Float16* __restrict__ Qpack, _Float16* __restrict__ Kpack,
                  _Float16* __restrict__ Vt,
                  float* __restrict__ q2, float* __restrict__ k2,
                  float* __restrict__ upd_coords) {
  const int t = blockIdx.x;
  const int tid = threadIdx.x;
  const float qscale = 0.25f;  // 1/sqrt(DH)

  // Q / K packs: 12 heads x 64 slots
  for (int s = tid; s < NHEAD * 64; s += 64) {
    int h = s >> 6, c = s & 63;
    float qv = 0.f, kv = 0.f;
    if (c < 16) {
      qv = proj_qs[t * 192 + h * 16 + c] * qscale;
      kv = proj_ks[t * 192 + h * 16 + c];
    } else if (c >= 32 && c < 44) {
      int pd = c - 32;
      float cd = coords[t * 3 + pd % 3];
      qv = proj_qp[t * 144 + h * 12 + pd] + cd;
      kv = proj_kp[t * 144 + h * 12 + pd] + cd;
    }
    size_t base = ((size_t)h * L_SEQ + t) * 64 + c;
    Qpack[base] = (_Float16)qv;
    Kpack[base] = (_Float16)kv;
  }

  // q2 / k2
  if (tid < NHEAD) {
    int h = tid;
    float s = 0.f;
    for (int pd = 0; pd < 12; ++pd) {
      float v = proj_qp[t * 144 + h * 12 + pd] + coords[t * 3 + pd % 3];
      s += v * v;
    }
    q2[h * L_SEQ + t] = s;
  } else if (tid < 2 * NHEAD) {
    int h = tid - NHEAD;
    float s = 0.f;
    for (int pd = 0; pd < 12; ++pd) {
      float v = proj_kp[t * 144 + h * 12 + pd] + coords[t * 3 + pd % 3];
      s += v * v;
    }
    k2[h * L_SEQ + t] = s;
  }

  // Transposed V pack (48 cols per head, zero-padded)
  for (int s = tid; s < NHEAD * 48; s += 64) {
    int h = s / 48, c = s % 48;
    float v = 0.f;
    if (c < 16)       v = proj_vs[t * 192 + h * 16 + c];
    else if (c < 40)  v = proj_vp[t * 288 + h * 24 + (c - 16)];
    Vt[((size_t)h * 48 + c) * L_SEQ + t] = (_Float16)v;
  }

  // updated_coords = v_p.mean over (h, p)  (96 samples per d)
  if (tid < 3) {
    float s = 0.f;
    for (int i = 0; i < NHEAD * PV; ++i) s += proj_vp[t * 288 + i * 3 + tid];
    upd_coords[t * 3 + tid] = s * (1.0f / 96.0f);
  }
}

// ---------------------------------------------------------------------------
// Flash-attention with point term. One wave per (head, 16-row q tile).
// logit = q_s.k_s/4 - 0.5*softplus(q2 + k2 - 2*q_p.k_p)
// Online softmax; P re-shaped C-layout -> A-layout via a 1KB LDS tile,
// then 3 WMMAs accumulate P @ [v_s | v_p] (48 cols, K = 32 keys/block).
// ---------------------------------------------------------------------------
__global__ void __launch_bounds__(32)
ipa_attention(const _Float16* __restrict__ Qpack, const _Float16* __restrict__ Kpack,
              const _Float16* __restrict__ Vt,
              const float* __restrict__ q2, const float* __restrict__ k2,
              _Float16* __restrict__ attn_out /* [1024][480] f16 */) {
  __shared__ __align__(16) _Float16 Ptile[16 * 32];

  const int lane = threadIdx.x & 31;
  const int hi = lane >> 4;
  const int nn = lane & 15;
  const int h = blockIdx.x >> 6;   // 12 heads
  const int qt = blockIdx.x & 63;  // 64 q tiles
  const int q0 = qt * 16;

  const _Float16* Qh = Qpack + (size_t)h * L_SEQ * 64;
  const _Float16* Kh = Kpack + (size_t)h * L_SEQ * 64;
  const _Float16* Vh = Vt + (size_t)h * 48 * L_SEQ;

  const v16h a_s = load_frag(Qh, q0, 64, 0);   // scalar features (16 + pad)
  const v16h a_p = load_frag(Qh, q0, 64, 32);  // point features (12 + pad)

  float q2row[8];
#pragma unroll
  for (int r = 0; r < 8; ++r) q2row[r] = q2[h * L_SEQ + q0 + r + 8 * hi];

  float m[8], l[8];
  v8f o0 = {}, o1 = {}, o2 = {};
#pragma unroll
  for (int r = 0; r < 8; ++r) { m[r] = -1e30f; l[r] = 0.f; }

  for (int kb = 0; kb < L_SEQ; kb += 32) {
    v8f logit0, logit1;
#pragma unroll
    for (int sub = 0; sub < 2; ++sub) {
      const int k0 = kb + sub * 16;
      v16h b_s = load_frag(Kh, k0, 64, 0);
      v16h b_p = load_frag(Kh, k0, 64, 32);
      v8f zero = {};
      v8f sc = __builtin_amdgcn_wmma_f32_16x16x32_f16(false, a_s, false, b_s,
                                                      (short)0, zero, false, false);
      v8f cr = __builtin_amdgcn_wmma_f32_16x16x32_f16(false, a_p, false, b_p,
                                                      (short)0, zero, false, false);
      const float kk = k2[h * L_SEQ + k0 + nn];
#pragma unroll
      for (int r = 0; r < 8; ++r) {
        float d2 = q2row[r] + kk - 2.0f * cr[r];
        float sp = fmaxf(d2, 0.f) + log1pf(expf(-fabsf(d2)));  // softplus
        float lg = sc[r] - 0.5f * sp;
        if (sub == 0) logit0[r] = lg; else logit1[r] = lg;
      }
    }

    // ---- online softmax over this 32-key block ----
    float fr[8];
#pragma unroll
    for (int r = 0; r < 8; ++r) {
      float t = fmaxf(logit0[r], logit1[r]);
#pragma unroll
      for (int mm = 1; mm < 16; mm <<= 1) t = fmaxf(t, __shfl_xor(t, mm, 32));
      float mn = fmaxf(m[r], t);
      fr[r] = expf(m[r] - mn);
      m[r] = mn;
    }
#pragma unroll
    for (int r = 0; r < 8; ++r) {
      float p0 = expf(logit0[r] - m[r]);
      float p1 = expf(logit1[r] - m[r]);
      float rs = p0 + p1;
#pragma unroll
      for (int mm = 1; mm < 16; mm <<= 1) rs += __shfl_xor(rs, mm, 32);
      l[r] = l[r] * fr[r] + rs;
      o0[r] *= fr[r]; o1[r] *= fr[r]; o2[r] *= fr[r];
      const int row = r + 8 * hi;  // C-layout row owned by this lane
      Ptile[row * 32 + nn]      = (_Float16)p0;
      Ptile[row * 32 + 16 + nn] = (_Float16)p1;
    }
    __syncthreads();  // single-wave WG: lowers to S_NOP, orders LDS

    v16h pa = load_frag(Ptile, 0, 32, 0);       // P as A operand, K=32 keys
    v16h b0 = load_frag(Vh, 0,  L_SEQ, kb);     // v_s cols 0..15
    v16h b1 = load_frag(Vh, 16, L_SEQ, kb);     // v_p cols 16..31
    v16h b2 = load_frag(Vh, 32, L_SEQ, kb);     // v_p cols 32..39 (+pad)
    o0 = __builtin_amdgcn_wmma_f32_16x16x32_f16(false, pa, false, b0, (short)0, o0, false, false);
    o1 = __builtin_amdgcn_wmma_f32_16x16x32_f16(false, pa, false, b1, (short)0, o1, false, false);
    o2 = __builtin_amdgcn_wmma_f32_16x16x32_f16(false, pa, false, b2, (short)0, o2, false, false);
    __syncthreads();
  }

  // ---- normalize and scatter into concat layout [q][480]:
  //      cols h*16+c = out_s ; cols 192 + h*24 + j = out_p
#pragma unroll
  for (int r = 0; r < 8; ++r) {
    const int row = q0 + r + 8 * hi;
    const float inv = 1.0f / l[r];
    attn_out[(size_t)row * 480 + h * 16 + nn]            = (_Float16)(o0[r] * inv);
    attn_out[(size_t)row * 480 + 192 + h * 24 + nn]      = (_Float16)(o1[r] * inv);
    if (nn < 8)
      attn_out[(size_t)row * 480 + 192 + h * 24 + 16 + nn] = (_Float16)(o2[r] * inv);
  }
}

// ---------------------------------------------------------------------------
// Host-side orchestration
// ---------------------------------------------------------------------------
static inline size_t align256(size_t x) { return (x + 255) & ~(size_t)255; }

extern "C" void kernel_launch(void* const* d_in, const int* in_sizes, int n_in,
                              void* d_out, int out_size, void* d_ws, size_t ws_size,
                              hipStream_t stream) {
  (void)in_sizes; (void)n_in; (void)out_size; (void)ws_size;
  const float* x      = (const float*)d_in[0];   // 1024x384
  const float* coords = (const float*)d_in[1];   // 1024x3
  const float* w_qs = (const float*)d_in[2];  const float* b_qs = (const float*)d_in[3];
  const float* w_ks = (const float*)d_in[4];  const float* b_ks = (const float*)d_in[5];
  const float* w_vs = (const float*)d_in[6];  const float* b_vs = (const float*)d_in[7];
  const float* w_qp = (const float*)d_in[8];  const float* b_qp = (const float*)d_in[9];
  const float* w_kp = (const float*)d_in[10]; const float* b_kp = (const float*)d_in[11];
  const float* w_vp = (const float*)d_in[12]; const float* b_vp = (const float*)d_in[13];
  const float* w_out = (const float*)d_in[14]; const float* b_out = (const float*)d_in[15];

  float* out = (float*)d_out;                       // 1024*384
  float* upd_coords = out + (size_t)L_SEQ * DIM;    // 1024*3

  // --- workspace carve-out ---
  char* ws = (char*)d_ws;
  size_t off = 0;
  auto alloc = [&](size_t bytes) { char* p = ws + off; off = align256(off + bytes); return p; };

  _Float16* xh     = (_Float16*)alloc((size_t)L_SEQ * DIM * 2);
  _Float16* wt_qs  = (_Float16*)alloc((size_t)192 * DIM * 2);
  _Float16* wt_ks  = (_Float16*)alloc((size_t)192 * DIM * 2);
  _Float16* wt_vs  = (_Float16*)alloc((size_t)192 * DIM * 2);
  _Float16* wt_qp  = (_Float16*)alloc((size_t)144 * DIM * 2);
  _Float16* wt_kp  = (_Float16*)alloc((size_t)144 * DIM * 2);
  _Float16* wt_vp  = (_Float16*)alloc((size_t)288 * DIM * 2);
  _Float16* wt_out = (_Float16*)alloc((size_t)DIM * 480 * 2);
  float* proj_qs = (float*)alloc((size_t)L_SEQ * 192 * 4);
  float* proj_ks = (float*)alloc((size_t)L_SEQ * 192 * 4);
  float* proj_vs = (float*)alloc((size_t)L_SEQ * 192 * 4);
  float* proj_qp = (float*)alloc((size_t)L_SEQ * 144 * 4);
  float* proj_kp = (float*)alloc((size_t)L_SEQ * 144 * 4);
  float* proj_vp = (float*)alloc((size_t)L_SEQ * 288 * 4);
  _Float16* Qpack = (_Float16*)alloc((size_t)NHEAD * L_SEQ * 64 * 2);
  _Float16* Kpack = (_Float16*)alloc((size_t)NHEAD * L_SEQ * 64 * 2);
  _Float16* Vt    = (_Float16*)alloc((size_t)NHEAD * 48 * L_SEQ * 2);
  float* q2 = (float*)alloc((size_t)NHEAD * L_SEQ * 4);
  float* k2 = (float*)alloc((size_t)NHEAD * L_SEQ * 4);
  _Float16* attn_o = (_Float16*)alloc((size_t)L_SEQ * 480 * 2);

  // --- 1) f16 packs ---
  {
    int n = L_SEQ * DIM;
    cvt_f32_to_f16<<<(n + 255) / 256, 256, 0, stream>>>(x, xh, n);
  }
  auto tr = [&](const float* w, _Float16* wt, int K, int N) {
    int n = K * N;
    transpose_to_f16<<<(n + 255) / 256, 256, 0, stream>>>(w, wt, K, N);
  };
  tr(w_qs, wt_qs, DIM, 192); tr(w_ks, wt_ks, DIM, 192); tr(w_vs, wt_vs, DIM, 192);
  tr(w_qp, wt_qp, DIM, 144); tr(w_kp, wt_kp, DIM, 144); tr(w_vp, wt_vp, DIM, 288);
  tr(w_out, wt_out, 480, DIM);

  // --- 2) projections: proj = x @ W + b (WMMA) ---
  auto gemm = [&](const _Float16* A, const _Float16* Bt, const float* bias,
                  float* C, int N, int K) {
    dim3 grid(L_SEQ / 16, N / 16);
    gemm_wmma_f16f32<<<grid, 32, 0, stream>>>(A, Bt, bias, C, N, K);
  };
  gemm(xh, wt_qs, b_qs, proj_qs, 192, DIM);
  gemm(xh, wt_ks, b_ks, proj_ks, 192, DIM);
  gemm(xh, wt_vs, b_vs, proj_vs, 192, DIM);
  gemm(xh, wt_qp, b_qp, proj_qp, 144, DIM);
  gemm(xh, wt_kp, b_kp, proj_kp, 144, DIM);
  gemm(xh, wt_vp, b_vp, proj_vp, 288, DIM);

  // --- 3) build attention operand packs + q2/k2 + updated_coords ---
  ipa_pack_epilogue<<<L_SEQ, 64, 0, stream>>>(proj_qs, proj_ks, proj_vs,
                                              proj_qp, proj_kp, proj_vp,
                                              coords, Qpack, Kpack, Vt,
                                              q2, k2, upd_coords);

  // --- 4) flash attention with point term ---
  ipa_attention<<<NHEAD * (L_SEQ / 16), 32, 0, stream>>>(Qpack, Kpack, Vt,
                                                         q2, k2, attn_o);

  // --- 5) output projection: out = concat @ w_out + b_out (WMMA) ---
  {
    dim3 grid(L_SEQ / 16, DIM / 16);
    gemm_wmma_f16f32<<<grid, 32, 0, stream>>>(attn_o, wt_out, b_out, out, DIM, 480);
  }
}
